// GatedLinearAttention_13391708028967
// MI455X (gfx1250) — matrix operand order repaired and checked by reading, not compiled, not run on hardware
//
#include <hip/hip_runtime.h>

// ---------------------------------------------------------------------------
// Gated Linear Attention forward for gfx1250 (MI455X).
// WMMA f16 compute + async global->LDS (ASYNCcnt) double-buffered GEMM tiles.
// ---------------------------------------------------------------------------

typedef __attribute__((ext_vector_type(16))) _Float16 v16h;
typedef __attribute__((ext_vector_type(8)))  _Float16 v8h;
typedef __attribute__((ext_vector_type(8)))  float    v8f;

#define TOKENS 8192
#define DMODEL 2048
#define NHEAD  8
#define NPROJ  6144   // q(1024) | k(1024) | v(2048) | g(2048)
#define GKW    1024   // H * dk

static __device__ __forceinline__ v8f vzero8() {
  v8f z;
#pragma unroll
  for (int i = 0; i < 8; ++i) z[i] = 0.f;
  return z;
}

// A-matrix fragment (16x32 f16): lane holds row m=lane%16; halves are
// contiguous K = kbA..kbA+15, kbA = (lane<16)?0:16.
static __device__ __forceinline__ v16h load_fragA(const _Float16* p, int kbA) {
  v8h lo = *(const v8h*)(p + kbA);
  v8h hi = *(const v8h*)(p + kbA + 8);
  v16h r;
#pragma unroll
  for (int i = 0; i < 8; ++i) { r[i] = lo[i]; r[i + 8] = hi[i]; }
  return r;
}

// B-matrix fragment (32x16 f16): lane holds column n=lane%16; halves are
// contiguous K = kbB..kbB+15, kbB = (lane<16)?0:16.  p points at a row of a
// K-contiguous LDS tile Bs[n][k].
static __device__ __forceinline__ v16h load_fragB(const _Float16* p, int kbB) {
  v8h lo = *(const v8h*)(p + kbB);
  v8h hi = *(const v8h*)(p + kbB + 8);
  v16h r;
#pragma unroll
  for (int i = 0; i < 8; ++i) { r[i] = lo[i]; r[i + 8] = hi[i]; }
  return r;
}

#define WMMA_F16(a, b, c) \
  __builtin_amdgcn_wmma_f32_16x16x32_f16(false, (a), false, (b), (short)0, (c), false, false)

// Async DMA: 16 bytes global -> LDS, tracked by ASYNCcnt (ISA 10.x / 15.18.3).
static __device__ __forceinline__ void async_load_b128(const void* g, const void* lds) {
  unsigned int l32 = (unsigned int)(unsigned long long)lds;  // LDS addr = addr[31:0]
  asm volatile("global_load_async_to_lds_b128 %0, %1, off"
               :: "v"(l32), "v"(g) : "memory");
}

static __device__ __forceinline__ void wait_async0() {
#if __has_builtin(__builtin_amdgcn_s_wait_asynccnt)
  __builtin_amdgcn_s_wait_asynccnt(0);
#else
  asm volatile("s_wait_asynccnt 0" ::: "memory");
#endif
}

// ---------------------------------------------------------------------------
// f32 -> f16 convert (x only)
// ---------------------------------------------------------------------------
__global__ __launch_bounds__(256) void cvt_f32_f16_kernel(
    const float* __restrict__ in, _Float16* __restrict__ out, int n) {
  int i = blockIdx.x * 256 + threadIdx.x;
  if (i < n) out[i] = (_Float16)in[i];
}

// Pack [Wq | Wk*scaling | Wv | Wg] TRANSPOSED into WallT (NPROJ x DMODEL, f16):
// WallT[c][r] = W(r, c).  K-contiguous rows make the GEMM B tiles async-DMA-able.
__global__ __launch_bounds__(256) void pack_w_kernel(
    const float* __restrict__ Wq, const float* __restrict__ Wk,
    const float* __restrict__ Wv, const float* __restrict__ Wg,
    _Float16* __restrict__ WallT) {
  int idx = blockIdx.x * 256 + threadIdx.x;
  if (idx >= DMODEL * NPROJ) return;
  int c = idx / DMODEL, r = idx % DMODEL;
  float v;
  if (c < 1024)       v = Wq[r * 1024 + c];
  else if (c < 2048)  v = Wk[r * 1024 + (c - 1024)] * 0.0625f;  // (D/H)^-0.5
  else if (c < 4096)  v = Wv[r * 2048 + (c - 2048)];
  else                v = Wg[r * 2048 + (c - 4096)];
  WallT[idx] = (_Float16)v;
}

// Wo transposed to f16: WohT[c][r] = Wo[r][c].
__global__ __launch_bounds__(256) void pack_woT_kernel(
    const float* __restrict__ Wo, _Float16* __restrict__ WohT) {
  int idx = blockIdx.x * 256 + threadIdx.x;
  if (idx >= DMODEL * DMODEL) return;
  int c = idx >> 11, r = idx & 2047;
  WohT[idx] = (_Float16)Wo[(size_t)r * DMODEL + c];
}

// ---------------------------------------------------------------------------
// WMMA GEMM: C(MxN) = A(MxK,f16) * BT(NxK,f16)^T, f32 accumulate.
// Block = 256 threads = 8 waves (4x2), wave tile 32x64, WG tile 128x128.
// Tiles fetched with global_load_async_to_lds_b128, double-buffered in LDS.
// ---------------------------------------------------------------------------
#define BM 128
#define BN 128
#define BK 32
#define PA 40   // As half-pitch (BK + 8)
#define PB 40   // Bs half-pitch (BK + 8); Bs stored [n][k]

template <bool OUT_F16>
__global__ __launch_bounds__(256) void gemm_f16_kernel(
    const _Float16* __restrict__ A, const _Float16* __restrict__ BT,
    void* __restrict__ C, int M, int N, int K) {
  __shared__ __align__(16) _Float16 As[2][BM * PA];
  __shared__ __align__(16) _Float16 Bs[2][BN * PB];
  const int tid  = threadIdx.x;
  const int lane = tid & 31;
  const int wid  = tid >> 5;
  const int wm   = wid & 3;
  const int wn   = wid >> 2;
  const int ln   = lane & 15;
  const int kbA  = (lane < 16) ? 0 : 16;
  const int kbB  = (lane < 16) ? 0 : 16;
  const int m0   = blockIdx.x * BM;
  const int n0   = blockIdx.y * BN;

  auto issue_tile = [&](int k0, _Float16* Ad, _Float16* Bd) {
#pragma unroll
    for (int p = 0; p < 2; ++p) {
      int idx = (p * 256 + tid) * 8;
      int r = idx >> 5, c = idx & 31;               // 128 rows x 32 halves
      async_load_b128(A + (size_t)(m0 + r) * K + (k0 + c), Ad + r * PA + c);
    }
#pragma unroll
    for (int p = 0; p < 2; ++p) {
      int idx = (p * 256 + tid) * 8;
      int n = idx >> 5, c = idx & 31;               // 128 n-rows x 32 halves
      async_load_b128(BT + (size_t)(n0 + n) * K + (k0 + c), Bd + n * PB + c);
    }
  };

  v8f acc[2][4];
#pragma unroll
  for (int i = 0; i < 2; ++i)
#pragma unroll
    for (int j = 0; j < 4; ++j) acc[i][j] = vzero8();

  issue_tile(0, As[0], Bs[0]);
  const int nk = K / BK;
  for (int t = 0; t < nk; ++t) {
    wait_async0();
    __syncthreads();
    if (t + 1 < nk) issue_tile((t + 1) * BK, As[(t + 1) & 1], Bs[(t + 1) & 1]);
    const _Float16* Ac = As[t & 1];
    const _Float16* Bc = Bs[t & 1];

    v16h af[2], bf[4];
#pragma unroll
    for (int i = 0; i < 2; ++i)
      af[i] = load_fragA(Ac + (wm * 32 + i * 16 + ln) * PA, kbA);
#pragma unroll
    for (int j = 0; j < 4; ++j)
      bf[j] = load_fragB(Bc + (wn * 64 + j * 16 + ln) * PB, kbB);
#pragma unroll
    for (int i = 0; i < 2; ++i)
#pragma unroll
      for (int j = 0; j < 4; ++j) acc[i][j] = WMMA_F16(af[i], bf[j], acc[i][j]);
  }

  const int rofs = (lane < 16) ? 0 : 8;
#pragma unroll
  for (int i = 0; i < 2; ++i) {
    int mb = m0 + wm * 32 + i * 16 + rofs;
#pragma unroll
    for (int j = 0; j < 4; ++j) {
      int nc = n0 + wn * 64 + j * 16 + ln;
#pragma unroll
      for (int r = 0; r < 8; ++r) {
        size_t off = (size_t)(mb + r) * N + nc;
        if (OUT_F16) ((_Float16*)C)[off] = (_Float16)acc[i][j][r];
        else         ((float*)C)[off]    = acc[i][j][r];
      }
    }
  }
}

// ---------------------------------------------------------------------------
// gk = log_sigmoid((x @ Wkg1) @ Wkg2 + bkg2) / 16, one block per token.
// ---------------------------------------------------------------------------
__global__ __launch_bounds__(256) void gk_proj_kernel(
    const float* __restrict__ x, const float* __restrict__ Wkg1,
    const float* __restrict__ Wkg2, const float* __restrict__ bkg2,
    float* __restrict__ gkb) {
  const int row = blockIdx.x;
  const int tid = threadIdx.x, lane = tid & 31, wid = tid >> 5;
  float part[16] = {};
  for (int k = tid; k < DMODEL; k += 256) {
    float xv = x[(size_t)row * DMODEL + k];
    const float* wr = Wkg1 + (size_t)k * 16;
#pragma unroll
    for (int c = 0; c < 16; ++c) part[c] += xv * wr[c];
  }
  __shared__ float red[16 * 8];
  __shared__ float t1[16];
#pragma unroll
  for (int c = 0; c < 16; ++c) {
    float a = part[c];
#pragma unroll
    for (int off = 16; off; off >>= 1) a += __shfl_xor(a, off);
    if (lane == 0) red[c * 8 + wid] = a;
  }
  __syncthreads();
  if (tid < 16) {
    float s = 0.f;
#pragma unroll
    for (int w = 0; w < 8; ++w) s += red[tid * 8 + w];
    t1[tid] = s;
  }
  __syncthreads();
  for (int j = tid; j < GKW; j += 256) {
    float z = bkg2[j];
#pragma unroll
    for (int c = 0; c < 16; ++c) z += t1[c] * Wkg2[c * GKW + j];
    float ls = fminf(z, 0.f) - log1pf(__expf(-fabsf(z)));  // stable logsigmoid
    gkb[(size_t)row * GKW + j] = ls * (1.f / 16.f);
  }
}

// ---------------------------------------------------------------------------
// Chunked GLA. Grid = 4(b) * 8(h) * 2(dv-half) = 64 WGs of 256 threads.
// Per 128-token chunk: async-DMA gk tile + cumsum in LDS, bake decays into
// f16 tiles, then three WMMA phases. State S (128x128 slab) persists f32.
// ---------------------------------------------------------------------------
#define PF 136  // f16 pitch (128 + 8)
#define PS 132  // f32 pitch (128 + 4)

__global__ __launch_bounds__(256) void gla_attn_kernel(
    const _Float16* __restrict__ QKVG, const float* __restrict__ gkbuf,
    float* __restrict__ oPre) {
  extern __shared__ char smem[];
  _Float16* qd   = (_Float16*)smem;          // [128][PF] q * exp(gkc)
  _Float16* kdec = qd + 128 * PF;            // [128][PF] k * exp(-gkc)
  _Float16* kdT  = kdec + 128 * PF;          // [d][j] k * exp(gkl - gkc), transposed
  _Float16* vT   = kdT + 128 * PF;           // [vcol][j] v, transposed
  float*    gkc  = (float*)(vT + 128 * PF);  // [128][PS] cumsum(gk)  (reused as ALds)
  _Float16* ALds = (_Float16*)gkc;           // [i][j] masked intra-chunk A (overlay)
  float*    St   = gkc + 128 * PS;           // [vcol][d] state, f32, persistent
  float*    gkl  = St + 128 * PS;            // [128] last-row cumsum

  const int bid  = blockIdx.x;
  const int vs   = bid & 1;
  const int h    = (bid >> 1) & 7;
  const int bb   = bid >> 4;
  const int tid  = threadIdx.x;
  const int lane = tid & 31;
  const int wid  = tid >> 5;
  const int wm   = wid & 3, wn = wid >> 2;
  const int ln   = lane & 15;
  const int kbA  = (lane < 16) ? 0 : 16;
  const int kbB  = (lane < 16) ? 0 : 16;
  const int rofs = (lane < 16) ? 0 : 8;

  for (int i = tid; i < 128 * PS; i += 256) St[i] = 0.f;

  for (int nt = 0; nt < 16; ++nt) {
    const int rb = bb * 2048 + nt * 128;  // global token row of chunk start
    __syncthreads();

    // --- phase 1: async-DMA gk chunk into LDS, inclusive cumsum per column ---
    for (int u = tid; u < 128 * 32; u += 256) {      // 16B granules
      int i = u >> 5, f = (u & 31) * 4;
      async_load_b128(gkbuf + (size_t)(rb + i) * GKW + h * 128 + f,
                      gkc + i * PS + f);
    }
    wait_async0();
    __syncthreads();
    if (tid < 128) {
      float run = 0.f;
      for (int i = 0; i < 128; ++i) { run += gkc[i * PS + tid]; gkc[i * PS + tid] = run; }
      gkl[tid] = run;
    }
    __syncthreads();

    // --- phase 2: decayed tiles (f16) ---
    const _Float16* Qg = QKVG + (size_t)rb * NPROJ + h * 128;
    const _Float16* Kg = QKVG + (size_t)rb * NPROJ + 1024 + h * 128;
    const _Float16* Vg = QKVG + (size_t)rb * NPROJ + 2048 + h * 256 + vs * 128;
    for (int idx = tid; idx < 128 * 128; idx += 256) {
      int i = idx >> 7, d = idx & 127;
      float gc = gkc[i * PS + d];
      float qv = (float)Qg[(size_t)i * NPROJ + d];
      float kv = (float)Kg[(size_t)i * NPROJ + d];
      qd[i * PF + d]   = (_Float16)(qv * __expf(gc));
      kdec[i * PF + d] = (_Float16)(kv * __expf(-gc));
      kdT[d * PF + i]  = (_Float16)(kv * __expf(gkl[d] - gc));
      vT[d * PF + i]   = Vg[(size_t)i * NPROJ + d];
    }
    __syncthreads();

    // --- phase 3: A = qd @ kdec^T, causal mask, store to ALds ---
    {
      v8f a[2][4];
#pragma unroll
      for (int i = 0; i < 2; ++i)
#pragma unroll
        for (int j = 0; j < 4; ++j) a[i][j] = vzero8();
#pragma unroll
      for (int k0 = 0; k0 < 128; k0 += 32) {
        v16h af[2], bf[4];
#pragma unroll
        for (int i = 0; i < 2; ++i)
          af[i] = load_fragA(qd + (wm * 32 + i * 16 + ln) * PF + k0, kbA);
#pragma unroll
        for (int j = 0; j < 4; ++j)
          bf[j] = load_fragB(kdec + (wn * 64 + j * 16 + ln) * PF + k0, kbB);
#pragma unroll
        for (int i = 0; i < 2; ++i)
#pragma unroll
          for (int j = 0; j < 4; ++j) a[i][j] = WMMA_F16(af[i], bf[j], a[i][j]);
      }
#pragma unroll
      for (int i = 0; i < 2; ++i) {
        int mb = wm * 32 + i * 16 + rofs;
#pragma unroll
        for (int j = 0; j < 4; ++j) {
          int nc = wn * 64 + j * 16 + ln;
#pragma unroll
          for (int r = 0; r < 8; ++r) {
            int ir = mb + r;
            ALds[ir * PF + nc] = (_Float16)((ir >= nc) ? a[i][j][r] : 0.f);
          }
        }
      }
    }
    __syncthreads();

    // --- phase 4: o = qd @ S + mask(A) @ v ---
    {
      v8f oc[2][4];
#pragma unroll
      for (int i = 0; i < 2; ++i)
#pragma unroll
        for (int j = 0; j < 4; ++j) oc[i][j] = vzero8();
      // inter-chunk: B from f32 state (convert while building fragment)
#pragma unroll
      for (int k0 = 0; k0 < 128; k0 += 32) {
        v16h af[2], bf[4];
#pragma unroll
        for (int i = 0; i < 2; ++i)
          af[i] = load_fragA(qd + (wm * 32 + i * 16 + ln) * PF + k0, kbA);
#pragma unroll
        for (int j = 0; j < 4; ++j) {
          const float* sr = St + (size_t)(wn * 64 + j * 16 + ln) * PS + k0;
          v16h t;
#pragma unroll
          for (int e = 0; e < 8; ++e) {
            t[e]     = (_Float16)sr[kbB + e];
            t[e + 8] = (_Float16)sr[kbB + 8 + e];
          }
          bf[j] = t;
        }
#pragma unroll
        for (int i = 0; i < 2; ++i)
#pragma unroll
          for (int j = 0; j < 4; ++j) oc[i][j] = WMMA_F16(af[i], bf[j], oc[i][j]);
      }
      // intra-chunk
#pragma unroll
      for (int k0 = 0; k0 < 128; k0 += 32) {
        v16h af[2], bf[4];
#pragma unroll
        for (int i = 0; i < 2; ++i)
          af[i] = load_fragA(ALds + (wm * 32 + i * 16 + ln) * PF + k0, kbA);
#pragma unroll
        for (int j = 0; j < 4; ++j)
          bf[j] = load_fragB(vT + (wn * 64 + j * 16 + ln) * PF + k0, kbB);
#pragma unroll
        for (int i = 0; i < 2; ++i)
#pragma unroll
          for (int j = 0; j < 4; ++j) oc[i][j] = WMMA_F16(af[i], bf[j], oc[i][j]);
      }
      const int cb = h * 256 + vs * 128;
#pragma unroll
      for (int i = 0; i < 2; ++i) {
        int mb = wm * 32 + i * 16 + rofs;
#pragma unroll
        for (int j = 0; j < 4; ++j) {
          int nc = cb + wn * 64 + j * 16 + ln;
#pragma unroll
          for (int r = 0; r < 8; ++r)
            oPre[(size_t)(rb + mb + r) * DMODEL + nc] = oc[i][j][r];
        }
      }
    }
    __syncthreads();

    // --- phase 5: S = diag(exp(gkl)) * S + kd^T @ v  (each wave owns its tile) ---
    {
      v8f sc[2][4];
#pragma unroll
      for (int i = 0; i < 2; ++i) {
        int db = wm * 32 + i * 16 + rofs;
#pragma unroll
        for (int j = 0; j < 4; ++j) {
          int vc = wn * 64 + j * 16 + ln;
#pragma unroll
          for (int r = 0; r < 8; ++r) {
            int d = db + r;
            sc[i][j][r] = St[(size_t)vc * PS + d] * __expf(gkl[d]);
          }
        }
      }
#pragma unroll
      for (int k0 = 0; k0 < 128; k0 += 32) {
        v16h af[2], bf[4];
#pragma unroll
        for (int i = 0; i < 2; ++i)
          af[i] = load_fragA(kdT + (wm * 32 + i * 16 + ln) * PF + k0, kbA);
#pragma unroll
        for (int j = 0; j < 4; ++j)
          bf[j] = load_fragB(vT + (wn * 64 + j * 16 + ln) * PF + k0, kbB);
#pragma unroll
        for (int i = 0; i < 2; ++i)
#pragma unroll
          for (int j = 0; j < 4; ++j) sc[i][j] = WMMA_F16(af[i], bf[j], sc[i][j]);
      }
#pragma unroll
      for (int i = 0; i < 2; ++i) {
        int db = wm * 32 + i * 16 + rofs;
#pragma unroll
        for (int j = 0; j < 4; ++j) {
          int vc = wn * 64 + j * 16 + ln;
#pragma unroll
          for (int r = 0; r < 8; ++r)
            St[(size_t)vc * PS + (db + r)] = sc[i][j][r];
        }
      }
    }
  }
}

// ---------------------------------------------------------------------------
// GroupNorm (LayerNorm over dv=256, no affine) + SiLU gate.  One block per
// (token, head); 256 threads = 256 channels.
// ---------------------------------------------------------------------------
__global__ __launch_bounds__(256) void norm_gate_kernel(
    const float* __restrict__ oPre, const _Float16* __restrict__ QKVG,
    const float* __restrict__ bg, _Float16* __restrict__ Gated) {
  const int t = blockIdx.x >> 3;
  const int h = blockIdx.x & 7;
  const int j = threadIdx.x;
  const int lane = j & 31, wid = j >> 5;
  float v = oPre[(size_t)t * DMODEL + h * 256 + j];
  float a = v, s = v * v;
#pragma unroll
  for (int off = 16; off; off >>= 1) { a += __shfl_xor(a, off); s += __shfl_xor(s, off); }
  __shared__ float s1[8], s2[8];
  if (lane == 0) { s1[wid] = a; s2[wid] = s; }
  __syncthreads();
  float sum = 0.f, sq = 0.f;
#pragma unroll
  for (int w = 0; w < 8; ++w) { sum += s1[w]; sq += s2[w]; }
  float mu  = sum * (1.f / 256.f);
  float var = sq * (1.f / 256.f) - mu * mu;
  float on  = (v - mu) * rsqrtf(var + 1e-5f);
  float gv  = (float)QKVG[(size_t)t * NPROJ + 4096 + h * 256 + j] + bg[h * 256 + j];
  float sg  = gv / (1.f + __expf(-gv));
  Gated[(size_t)t * DMODEL + h * 256 + j] = (_Float16)(sg * on);
}

// ---------------------------------------------------------------------------
// Host launcher
// ---------------------------------------------------------------------------
extern "C" void kernel_launch(void* const* d_in, const int* in_sizes, int n_in,
                              void* d_out, int out_size, void* d_ws, size_t ws_size,
                              hipStream_t stream) {
  (void)in_sizes; (void)n_in; (void)out_size; (void)ws_size;
  const float* x    = (const float*)d_in[0];
  const float* Wq   = (const float*)d_in[1];
  const float* Wk   = (const float*)d_in[2];
  const float* Wkg1 = (const float*)d_in[3];
  const float* Wkg2 = (const float*)d_in[4];
  const float* bkg2 = (const float*)d_in[5];
  const float* Wv   = (const float*)d_in[6];
  const float* Wg   = (const float*)d_in[7];
  const float* bg   = (const float*)d_in[8];
  const float* Wo   = (const float*)d_in[9];
  float* out = (float*)d_out;

  char* p = (char*)d_ws;
  auto take = [&](size_t bytes) -> char* {
    char* q = p; p += (bytes + 255) & ~(size_t)255; return q;
  };
  _Float16* Xh    = (_Float16*)take((size_t)TOKENS * DMODEL * 2);
  _Float16* WallT = (_Float16*)take((size_t)DMODEL * NPROJ * 2);
  _Float16* WohT  = (_Float16*)take((size_t)DMODEL * DMODEL * 2);
  _Float16* QKVG  = (_Float16*)take((size_t)TOKENS * NPROJ * 2);
  float*    gkb   = (float*)take((size_t)TOKENS * GKW * 4);
  float*    oPre  = (float*)take((size_t)TOKENS * DMODEL * 4);
  _Float16* Gated = (_Float16*)take((size_t)TOKENS * DMODEL * 2);

  // 1) precision packing (weights pre-transposed to [N][K] for async B tiles)
  cvt_f32_f16_kernel<<<(TOKENS * DMODEL) / 256, 256, 0, stream>>>(x, Xh, TOKENS * DMODEL);
  pack_woT_kernel<<<(DMODEL * DMODEL) / 256, 256, 0, stream>>>(Wo, WohT);
  pack_w_kernel<<<(DMODEL * NPROJ) / 256, 256, 0, stream>>>(Wq, Wk, Wv, Wg, WallT);

  // 2) gk path (small, f32)
  gk_proj_kernel<<<TOKENS, 256, 0, stream>>>(x, Wkg1, Wkg2, bkg2, gkb);

  // 3) fused q/k/v/g projection (WMMA + async double-buffered tiles)
  gemm_f16_kernel<true><<<dim3(TOKENS / 128, NPROJ / 128), 256, 0, stream>>>(
      Xh, WallT, (void*)QKVG, TOKENS, NPROJ, DMODEL);

  // 4) chunked GLA scan (WMMA, dynamic LDS ~275KB of the 320KB/WGP)
  constexpr size_t ATTN_SMEM = (size_t)4 * 128 * PF * sizeof(_Float16)
                             + (size_t)2 * 128 * PS * sizeof(float)
                             + 256 * sizeof(float);
  (void)hipFuncSetAttribute((const void*)gla_attn_kernel,
                            hipFuncAttributeMaxDynamicSharedMemorySize,
                            (int)ATTN_SMEM);
  gla_attn_kernel<<<64, 256, ATTN_SMEM, stream>>>(QKVG, gkb, oPre);

  // 5) group-norm + SiLU gate
  norm_gate_kernel<<<TOKENS * NHEAD, 256, 0, stream>>>(oPre, QKVG, bg, Gated);

  // 6) output projection (WMMA, f32 out)
  gemm_f16_kernel<false><<<dim3(TOKENS / 128, DMODEL / 128), 256, 0, stream>>>(
      Gated, WohT, (void*)out, TOKENS, DMODEL, DMODEL);
}
